// GDN_84456236908666
// MI455X (gfx1250) — matrix operand order repaired and checked by reading, not compiled, not run on hardware
//
#include <hip/hip_runtime.h>

typedef _Float16 v16h __attribute__((ext_vector_type(16)));
typedef _Float16 v8h  __attribute__((ext_vector_type(8)));
typedef _Float16 v4h  __attribute__((ext_vector_type(4)));
typedef float    v8f  __attribute__((ext_vector_type(8)));

#define CCH 192          // channels
#define HW  65536        // 256*256 pixels per (batch, channel) plane
#define NT  128          // pixel tile per workgroup
#define CP  200          // padded channel stride in LDS (f16 units): 400 B rows
                         //  -> scatter-store bank step 36 mod 64 (2-way max)

// ---------------------------------------------------------------------------
// Kernel 0: gamma (f32 [192][192]) -> f16, pre-swizzled into the exact
// per-lane VGPR order of the WMMA 16-bit A fragment (ISA 7.12.2):
//   lanes 0-15 : row M = lane,    halves 0..7 -> K = 0..7,  halves 8..15 -> K = 16..23
//   lanes 16-31: row M = lane-16, halves 0..7 -> K = 8..15, halves 8..15 -> K = 24..31
// Fragment (ot,k) occupies a contiguous 1 KB block: 32 lanes x 16 halves.
// ---------------------------------------------------------------------------
__global__ void gdn_gamma_swizzle(const float* __restrict__ g,
                                  _Float16* __restrict__ gsw) {
  int idx = blockIdx.x * 256 + threadIdx.x;      // 12*6*32*16 = 36864 total
  if (idx >= 12 * 6 * 32 * 16) return;
  int h    = idx & 15;
  int l    = (idx >> 4) & 31;
  int frag = idx >> 9;                           // ot*6 + k
  int k    = frag % 6;
  int ot   = frag / 6;
  int o    = ot * 16 + (l & 15);
  int base = (l >> 4) * 8;
  int krel = (h < 8) ? (base + h) : (16 + base + (h - 8));
  int c    = k * 32 + krel;
  gsw[idx] = (_Float16)g[o * CCH + c];
}

// ---------------------------------------------------------------------------
// Kernel 1: main GDN.  Block = 256 threads (8 wave32), one 128-pixel tile.
//   fill:    x^2 -> f16 into LDS pixel-major [NT][CP] (coalesced global reads)
//   compute: wave w fixes pixel sub-tile n0 = 16*w, keeps all 6 B fragments
//            in registers, sweeps 12 Cout tiles x 6 K-steps of
//            v_wmma_f32_16x16x32_f16
//   epilogue: out = x * rsqrt(beta + mix), non-temporal stores
// ---------------------------------------------------------------------------
__global__ __launch_bounds__(256, 2)
void gdn_main(const float* __restrict__ x, const float* __restrict__ beta,
              const _Float16* __restrict__ gsw, float* __restrict__ out) {
  __shared__ _Float16 xsT[NT * CP];              // 51200 B

  const int t   = threadIdx.x;
  const int blk = blockIdx.x;                    // 8 batches * 512 tiles
  const int b   = blk >> 9;
  const int p0  = (blk & 511) * NT;

  // ---- fill: square + cvt to f16, transpose into pixel-major LDS ----
  {
    const int p  = t & 127;                      // pixel in tile
    const int c0 = t >> 7;                       // 0 or 1
    const float* xp = x + (size_t)b * CCH * HW + p0 + p;
    _Float16* dst = &xsT[p * CP];
    #pragma unroll 4
    for (int c = c0; c < CCH; c += 2) {
      float v = xp[(size_t)c * HW];              // coalesced along p
      dst[c] = (_Float16)(v * v);                // 2-way-max bank conflicts
    }
  }
  __syncthreads();

  const int lane = t & 31;
  const int w    = t >> 5;                       // wave id 0..7
  const int n0   = w * 16;                       // pixel sub-tile
  const int ln   = lane & 15;
  const int hi16 = (lane >> 4) * 8;              // lanes 16-31 hold K+8..15, K+24..31

  // ---- load all 6 B fragments (x^2) once; two 16 B runs, 32 B apart, per lane ----
  v16h bf[6];
  #pragma unroll
  for (int k = 0; k < 6; ++k) {
    const v4h* p4 = (const v4h*)&xsT[(n0 + ln) * CP + k * 32 + hi16];
    union { v16h v; v4h q[4]; } u;
    u.q[0] = p4[0]; u.q[1] = p4[1]; u.q[2] = p4[4]; u.q[3] = p4[5];
    bf[k] = u.v;
  }

  const int rowhalf = (lane >> 4) * 8;           // D: lanes 16-31 hold M+8

  for (int ot = 0; ot < 12; ++ot) {
    v8f acc = {};
    #pragma unroll
    for (int k = 0; k < 6; ++k) {
      // A fragment: pre-swizzled, contiguous 32 B per lane from global/L2
      const v8h* ga = (const v8h*)(gsw + ((size_t)((ot * 6 + k) * 32 + lane) << 4));
      union { v16h v; v8h q[2]; } ua;
      ua.q[0] = ga[0]; ua.q[1] = ga[1];
      acc = __builtin_amdgcn_wmma_f32_16x16x32_f16(
          /*neg_a=*/false, ua.v, /*neg_b=*/false, bf[k],
          /*c_mod=*/(short)0, acc, /*reuse_a=*/false, /*reuse_b=*/false);
    }
    // ---- epilogue: D VGPR r -> (o = ot*16 + rowhalf + r, n = n0 + ln) ----
    const int obase = ot * 16 + rowhalf;
    #pragma unroll
    for (int r = 0; r < 8; ++r) {
      int o = obase + r;
      size_t idx = ((size_t)b * CCH + o) * HW + p0 + n0 + ln;
      float xv = x[idx];                         // cache-hot reload
      float bt = beta[o];
      float yv = xv * rsqrtf(bt + acc[r]);
      __builtin_nontemporal_store(yv, &out[idx]); // streamed, bypass reuse
    }
  }
}

extern "C" void kernel_launch(void* const* d_in, const int* in_sizes, int n_in,
                              void* d_out, int out_size, void* d_ws, size_t ws_size,
                              hipStream_t stream) {
  const float* x     = (const float*)d_in[0];
  const float* beta  = (const float*)d_in[1];
  const float* gamma = (const float*)d_in[2];
  float* out = (float*)d_out;
  _Float16* gsw = (_Float16*)d_ws;               // 73728 B of workspace

  gdn_gamma_swizzle<<<144, 256, 0, stream>>>(gamma, gsw);
  gdn_main<<<8 * 512, 256, 0, stream>>>(x, beta, gsw, out);
}
